// BiLSTMClassifier_11063835755285
// MI455X (gfx1250) — matrix operand order repaired and checked by reading, not compiled, not run on hardware
//
#include <hip/hip_runtime.h>

typedef unsigned short u16;
typedef unsigned int   u32;
typedef unsigned long long u64;

typedef __attribute__((ext_vector_type(16))) __bf16 v16bf;
typedef __attribute__((ext_vector_type(8)))  float  v8f;
typedef __attribute__((ext_vector_type(4)))  u32    v4u;
typedef __attribute__((ext_vector_type(8)))  int    v8i;
typedef __attribute__((ext_vector_type(4)))  int    v4i;

union Frag32B { uint4 u4[2]; v16bf v; };

#define BSZ   256
#define TLEN  512
#define EMB   128
#define HID   256
#define KTOT  384
#define NBLK  64            // 2 dirs * 8 col-groups * 4 row-groups

// LDS pitches (elements). TDM pads every 64/128 DWORDs by 4 DWORDs (8 elems):
#define WP    408           // 384 + 3*8   (pad_interval=64 dw, 3 pads/row)  816B = 51*16
#define AEP   136           // emb tile pitch (manual)                        272B = 17*16
#define AHP   264           // 256 + 8     (pad_interval=128 dw, 1 pad/row)  528B = 33*16

// workspace byte offsets (16B aligned)
#define OFF_BAR   0ull
#define OFF_BIAS  256ull
#define OFF_H     8448ull        // 2 bufs * 2 dir * 256*256 bf16 = 524288 B
#define OFF_WT    532736ull      // 2 dir * 1024 * 384 bf16 = 1572864 B
#define OFF_EMB   2105600ull     // 32000*128 bf16 = 8192000 B
#define OFF_FEAT  10297600ull    // 256*512 f32 = 524288 B

#if defined(__has_builtin)
#if __has_builtin(__builtin_amdgcn_tensor_load_to_lds)
#define HAVE_TDM 1
#endif
#endif
#if defined(__has_include)
#if __has_include(<hip/amd_detail/amd_gfx1250_TDM.h>)
#define TDM_6ARG 1            // therock toolchain: 6-arg builtin
#endif
#endif

__device__ __forceinline__ u16 f2bf(float f) {
  u32 u = __float_as_uint(f);
  u32 r = u + 0x7FFFu + ((u >> 16) & 1u);
  return (u16)(r >> 16);
}
__device__ __forceinline__ float sigmoidf(float x) { return 1.0f / (1.0f + __expf(-x)); }

#if defined(HAVE_TDM)
__device__ __forceinline__ u32 lds_off(const void* p) {
  return (u32)(uintptr_t)(__attribute__((address_space(3))) const void*)p;
}
__device__ __forceinline__ void tdm_load(v4u g0, v8i g1, v4i g2, v4i g3) {
#if defined(TDM_6ARG)
  v8i g4 = {0, 0, 0, 0, 0, 0, 0, 0};
  __builtin_amdgcn_tensor_load_to_lds(g0, g1, g2, g3, g4, 0);
#else
  __builtin_amdgcn_tensor_load_to_lds(g0, g1, g2, g3, 0);
#endif
}
__device__ __forceinline__ v4u tdm_g0(u32 ldsoff, u64 gaddr) {
  v4u g0;
  g0.x = 1u;                                 // count=1, user descriptor
  g0.y = ldsoff;                             // lds_addr (bytes)
  g0.z = (u32)gaddr;                         // global_addr[31:0]
  g0.w = (u32)(gaddr >> 32) | (2u << 30);    // global_addr[56:32] | type=2
  return g0;
}
#endif

// ---------------------------------------------------------------- prep kernels

__global__ void k_prep_small(const float* __restrict__ bf, const float* __restrict__ bb,
                             u32* __restrict__ bar, float* __restrict__ bias,
                             u32* __restrict__ hzero) {
  int i = blockIdx.x * blockDim.x + threadIdx.x;   // 512*256 = 131072 threads
  if (i < 64) bar[i] = 0u;
  if (i < 131072) hzero[i] = 0u;
  if (i < 1024)      bias[i] = bf[i];
  else if (i < 2048) bias[i] = bb[i - 1024];
}

__global__ void k_embT(const float* __restrict__ E, u16* __restrict__ out, int n) {
  for (int i = blockIdx.x * blockDim.x + threadIdx.x; i < n; i += gridDim.x * blockDim.x)
    out[i] = f2bf(E[i]);
}

// WtG[dir][n][k] = (k<128 ? Wx[k][n] : Wh[k-128][n]) bf16, N-major
__global__ void k_prepw(const float* __restrict__ Wxf, const float* __restrict__ Whf,
                        const float* __restrict__ Wxb, const float* __restrict__ Whb,
                        u16* __restrict__ WtG) {
  int i = blockIdx.x * blockDim.x + threadIdx.x;   // 3072*256 = 786432 exact
  if (i >= 2 * 1024 * KTOT) return;
  int dir = i / (1024 * KTOT);
  int rem = i - dir * 1024 * KTOT;
  int n = rem / KTOT, k = rem - n * KTOT;
  const float* Wx = dir ? Wxb : Wxf;
  const float* Wh = dir ? Whb : Whf;
  float v = (k < EMB) ? Wx[k * 1024 + n] : Wh[(k - EMB) * 1024 + n];
  WtG[i] = f2bf(v);
}

// ---------------------------------------------------------------- grid barrier

__device__ __forceinline__ void grid_barrier(u32* bar) {
  __threadfence();
  __syncthreads();
  if (threadIdx.x == 0) {
    u32 g = __hip_atomic_load(&bar[1], __ATOMIC_RELAXED, __HIP_MEMORY_SCOPE_AGENT);
    u32 a = __hip_atomic_fetch_add(&bar[0], 1u, __ATOMIC_ACQ_REL, __HIP_MEMORY_SCOPE_AGENT);
    if (a == (u32)(NBLK - 1)) {
      __hip_atomic_store(&bar[0], 0u,     __ATOMIC_RELAXED, __HIP_MEMORY_SCOPE_AGENT);
      __hip_atomic_store(&bar[1], g + 1u, __ATOMIC_RELEASE, __HIP_MEMORY_SCOPE_AGENT);
    } else {
      while (__hip_atomic_load(&bar[1], __ATOMIC_ACQUIRE, __HIP_MEMORY_SCOPE_AGENT) == g)
        __builtin_amdgcn_s_sleep(1);
    }
  }
  __syncthreads();
}

// ---------------------------------------------------------------- persistent BiLSTM

__global__ void __launch_bounds__(256, 1)
k_bilstm(const int* __restrict__ x, const u16* __restrict__ embT,
         const u16* __restrict__ WtG, const float* __restrict__ bias,
         u16* __restrict__ hbuf, float* __restrict__ feat, u32* __restrict__ bar) {
  __shared__ u16 sWt[128 * WP];    // [gate*32 + ct*16 + n][k], pad every 128 elems
  __shared__ u16 sAe[64 * AEP];    // emb half of A
  __shared__ u16 sAh[64 * AHP];    // h half of A, pad every 128 elems

  const int bid  = blockIdx.x;
  const int dir  = bid & 1;
  const int rest = bid >> 1;
  const int g    = rest & 7;
  const int r    = rest >> 3;
  const int hc0  = g * 32;
  const int r0   = r * 64;

  const int tid  = threadIdx.x;
  const int lane = tid & 31;
  const int wave = tid >> 5;
  const int rt   = wave >> 1;
  const int ct   = wave & 1;
  const int m    = lane & 15;
  const int hf   = lane >> 4;

  // ---- weight slice into LDS (once): 4 gate-chunks of 32 rows x 384 K
#if defined(HAVE_TDM)
  if (wave == 0) {
    u64 ga = (u64)(uintptr_t)WtG + ((u64)dir * 1024 + (u64)hc0) * (KTOT * 2);
    // data_size=2B, pad_en, pad_interval=64dw(code5), pad_amount=4dw(code3)
    v8i g1 = { (int)0x07510000,
               (int)0x01800000,   // tensor_dim0 = 384 (bits 79:48, low half)
               (int)0x04000000,   // tensor_dim1 = 1024
               (int)0x01800000,   // tile_dim0 = 384
               (int)0x00040020,   // tile_dim1 = 32 | tile_dim2 = 4
               (int)384,          // tensor_dim0_stride lo
               (int)0x80000000,   // tensor_dim1_stride (98304) low 16 in [31:16]
               (int)1 };          // tensor_dim1_stride high bits
    v4i g2 = {4, 0, 0, 0};        // tensor_dim2 = 4
    v4i g3 = {0, 0, 0, 0};
    tdm_load(tdm_g0(lds_off(sWt), ga), g1, g2, g3);
    __builtin_amdgcn_s_wait_tensorcnt(0);
  }
#else
  for (int nl = tid; nl < 128; nl += blockDim.x) {
    int gate = nl >> 5, j = nl & 31;
    int gn = gate * 256 + hc0 + j;
    const uint4* src = (const uint4*)(WtG + ((size_t)dir * 1024 + gn) * KTOT);
#pragma unroll
    for (int q = 0; q < 48; ++q)
      *(uint4*)(&sWt[nl * WP + q * 8 + (q >> 4) * 8]) = src[q];
  }
#endif

  float biasv[4];
#pragma unroll
  for (int gate = 0; gate < 4; ++gate)
    biasv[gate] = bias[dir * 1024 + gate * 256 + hc0 + ct * 16 + m];

  v8f creg = {0.f, 0.f, 0.f, 0.f, 0.f, 0.f, 0.f, 0.f};
  __syncthreads();

#pragma unroll 1
  for (int t = 0; t < TLEN; ++t) {
    const int rd = t & 1, wr = rd ^ 1;

    // ---- async DMA of h_{t-1} tile (64 contiguous rows x 256 bf16) into sAh
#if defined(HAVE_TDM)
    if (wave == 0) {
      u64 ga = (u64)(uintptr_t)hbuf + ((u64)(rd * 2 + dir) * BSZ + (u64)r0) * (HID * 2);
      // data_size=2B, pad_en, pad_interval=128dw(code6), pad_amount=4dw(code3)
      v8i g1 = { (int)0x07910000,
                 (int)0x01000000,   // tensor_dim0 = 256
                 (int)0x00400000,   // tensor_dim1 = 64
                 (int)0x01000000,   // tile_dim0 = 256
                 (int)0x00000040,   // tile_dim1 = 64
                 (int)256,          // tensor_dim0_stride
                 0, 0 };
      v4i gz = {0, 0, 0, 0};
      tdm_load(tdm_g0(lds_off(sAh), ga), g1, gz, gz);
    }
#endif

    // ---- stage embedding half of A (token gather, 4 threads/row)
    {
      const int r_l = tid >> 2;
      const int p   = tid & 3;
      const int b   = r0 + r_l;
      const int tt  = dir ? (TLEN - 1 - t) : t;
      const int tok = x[(size_t)b * TLEN + tt];
      const uint4* esrc = (const uint4*)(embT + (size_t)tok * EMB);
#pragma unroll
      for (int q = 0; q < 4; ++q) {
        int col = p * 32 + q * 8;
        *(uint4*)(&sAe[r_l * AEP + col]) = esrc[col >> 3];
      }
#if !defined(HAVE_TDM)
      const uint4* hsrc = (const uint4*)(hbuf + ((size_t)(rd * 2 + dir) * BSZ + b) * HID);
#pragma unroll
      for (int q = 0; q < 8; ++q) {
        int col = p * 64 + q * 8;
        *(uint4*)(&sAh[r_l * AHP + col + ((col >> 7) * 8)]) = hsrc[col >> 3];
      }
#endif
      if (t + 1 < TLEN) {
        int tt2 = dir ? (TLEN - 2 - t) : (t + 1);
        int tok2 = x[(size_t)b * TLEN + tt2];
        __builtin_prefetch(embT + (size_t)tok2 * EMB, 0, 1);
      }
    }
#if defined(HAVE_TDM)
    if (wave == 0) __builtin_amdgcn_s_wait_tensorcnt(0);
#endif
    __syncthreads();

    // ---- Z = [emb | h] @ W for our (16-row,16-hcol) tile, 4 gates, K=384
    v8f zi = {0.f,0.f,0.f,0.f,0.f,0.f,0.f,0.f};
    v8f zf = zi, zg = zi, zo = zi;
    const int arow = rt * 16 + m;
    const int nr0 = (0 * 32 + ct * 16 + m) * WP;
    const int nr1 = (1 * 32 + ct * 16 + m) * WP;
    const int nr2 = (2 * 32 + ct * 16 + m) * WP;
    const int nr3 = (3 * 32 + ct * 16 + m) * WP;

#pragma unroll
    for (int ks = 0; ks < 12; ++ks) {
      const int k0 = ks * 32;
      Frag32B a;
      if (ks < 4) {
        const u16* ar = &sAe[arow * AEP + k0 + hf * 8];
        a.u4[0] = *(const uint4*)(ar);
        a.u4[1] = *(const uint4*)(ar + 16);
      } else {
        const int c0 = (k0 - EMB) + hf * 8;
        const int c1 = c0 + 16;
        a.u4[0] = *(const uint4*)(&sAh[arow * AHP + c0 + ((c0 >> 7) * 8)]);
        a.u4[1] = *(const uint4*)(&sAh[arow * AHP + c1 + ((c1 >> 7) * 8)]);
      }
      const int koff = k0 + hf * 8;
      const int kpad = koff + ((koff >> 7) * 8);
      Frag32B b0, b1, b2, b3;
      const u16* p0 = &sWt[nr0 + kpad];
      const u16* p1 = &sWt[nr1 + kpad];
      const u16* p2 = &sWt[nr2 + kpad];
      const u16* p3 = &sWt[nr3 + kpad];
      b0.u4[0] = *(const uint4*)(p0); b0.u4[1] = *(const uint4*)(p0 + 16);
      b1.u4[0] = *(const uint4*)(p1); b1.u4[1] = *(const uint4*)(p1 + 16);
      b2.u4[0] = *(const uint4*)(p2); b2.u4[1] = *(const uint4*)(p2 + 16);
      b3.u4[0] = *(const uint4*)(p3); b3.u4[1] = *(const uint4*)(p3 + 16);

      zi = __builtin_amdgcn_wmma_f32_16x16x32_bf16(false, a.v, false, b0.v, (short)0, zi, false, false);
      zf = __builtin_amdgcn_wmma_f32_16x16x32_bf16(false, a.v, false, b1.v, (short)0, zf, false, false);
      zg = __builtin_amdgcn_wmma_f32_16x16x32_bf16(false, a.v, false, b2.v, (short)0, zg, false, false);
      zo = __builtin_amdgcn_wmma_f32_16x16x32_bf16(false, a.v, false, b3.v, (short)0, zo, false, false);
    }

    // ---- gates + state update; write h_t patch (bf16)
    u16* hout = hbuf + (size_t)(wr * 2 + dir) * BSZ * HID;
#pragma unroll
    for (int j = 0; j < 8; ++j) {
      float iv = sigmoidf(zi[j] + biasv[0]);
      float fv = sigmoidf(zf[j] + biasv[1]);
      float gv = tanhf   (zg[j] + biasv[2]);
      float ov = sigmoidf(zo[j] + biasv[3]);
      float cn = fv * creg[j] + iv * gv;
      creg[j] = cn;
      float hn = ov * tanhf(cn);
      int mrow = r0 + rt * 16 + j + 8 * hf;
      int ncol = hc0 + ct * 16 + m;
      hout[(size_t)mrow * HID + ncol] = f2bf(hn);
    }

    grid_barrier(bar);
  }

  // ---- final cell state -> feature buffer [256][512] (fwd | bwd)
#pragma unroll
  for (int j = 0; j < 8; ++j) {
    int mrow = r0 + rt * 16 + j + 8 * hf;
    int ncol = hc0 + ct * 16 + m;
    feat[(size_t)mrow * (2 * HID) + dir * HID + ncol] = creg[j];
  }
}

// ---------------------------------------------------------------- classifier

__global__ void k_cls(const float* __restrict__ feat, const float* __restrict__ Wd,
                      const float* __restrict__ bd, float* __restrict__ out) {
  int b = threadIdx.x;
  float a0 = bd[0], a1 = bd[1], a2 = bd[2], a3 = bd[3];
  for (int k = 0; k < 2 * HID; ++k) {
    float f = feat[(size_t)b * (2 * HID) + k];
    a0 += f * Wd[k * 4 + 0];
    a1 += f * Wd[k * 4 + 1];
    a2 += f * Wd[k * 4 + 2];
    a3 += f * Wd[k * 4 + 3];
  }
  out[b * 4 + 0] = a0; out[b * 4 + 1] = a1;
  out[b * 4 + 2] = a2; out[b * 4 + 3] = a3;
}

// ---------------------------------------------------------------- launch

extern "C" void kernel_launch(void* const* d_in, const int* in_sizes, int n_in,
                              void* d_out, int out_size, void* d_ws, size_t ws_size,
                              hipStream_t stream) {
  const int*   x   = (const int*)  d_in[0];
  const float* emb = (const float*)d_in[2];
  const float* Wxf = (const float*)d_in[3];
  const float* Whf = (const float*)d_in[4];
  const float* bf  = (const float*)d_in[5];
  const float* Wxb = (const float*)d_in[6];
  const float* Whb = (const float*)d_in[7];
  const float* bb  = (const float*)d_in[8];
  const float* Wd  = (const float*)d_in[9];
  const float* bd  = (const float*)d_in[10];

  char* ws = (char*)d_ws;
  u32*   bar  = (u32*)  (ws + OFF_BAR);
  float* bias = (float*)(ws + OFF_BIAS);
  u16*   hbuf = (u16*)  (ws + OFF_H);
  u16*   WtG  = (u16*)  (ws + OFF_WT);
  u16*   embT = (u16*)  (ws + OFF_EMB);
  float* feat = (float*)(ws + OFF_FEAT);

  k_prep_small<<<512, 256, 0, stream>>>(bf, bb, bar, bias, (u32*)hbuf);
  k_embT<<<2048, 256, 0, stream>>>(emb, embT, 32000 * EMB);
  k_prepw<<<3072, 256, 0, stream>>>(Wxf, Whf, Wxb, Whb, WtG);
  k_bilstm<<<NBLK, 256, 0, stream>>>(x, embT, WtG, bias, hbuf, feat, bar);
  k_cls<<<1, 256, 0, stream>>>(feat, Wd, bd, (float*)d_out);
}
